// BiLstmCRF_82557861364105
// MI455X (gfx1250) — matrix operand order repaired and checked by reading, not compiled, not run on hardware
//
#include <hip/hip_runtime.h>
#include <hip/hip_bf16.h>

typedef __bf16 bf16;
typedef __attribute__((ext_vector_type(8)))  bf16  v8bf;
typedef __attribute__((ext_vector_type(16))) bf16  v16bf;
typedef __attribute__((ext_vector_type(8)))  float v8f;

// ---------- WMMA helpers ----------
__device__ inline v16bf cat8(v8bf lo, v8bf hi) {
  v16bf r;
#pragma unroll
  for (int i = 0; i < 8; i++) { r[i] = lo[i]; r[i + 8] = hi[i]; }
  return r;
}

// A fragment (16x32 bf16, row-major, lane m = lane&15):
// lanes 0-15: e0..15 -> K=k0..k0+15
// lanes16-31: e0..15 -> K=k0+16..k0+31
__device__ inline v16bf loadA_frag(const bf16* rowPtrK0, int hi) {
  v8bf lo = *(const v8bf*)(rowPtrK0 + 16 * hi);
  v8bf hh = *(const v8bf*)(rowPtrK0 + 16 * hi + 8);
  return cat8(lo, hh);
}

// B fragment (32x16 bf16): lane n = lane&15 selects column n of K x N
// (== row n of the N x K weight, contiguous).  lanes<16: K=k0..15, lanes>=16: K=k0+16..31
__device__ inline v16bf loadB_frag(const bf16* rowPtrK0, int hi) {
  v8bf lo = *(const v8bf*)(rowPtrK0 + 16 * hi);
  v8bf hh = *(const v8bf*)(rowPtrK0 + 16 * hi + 8);
  return cat8(lo, hh);
}

__device__ inline v8f wmma_bf16(v16bf a, v16bf b, v8f c) {
  return __builtin_amdgcn_wmma_f32_16x16x32_bf16(false, a, false, b, (short)0, c,
                                                 false, false);
}

__device__ inline v8f vzero8() {
  v8f z;
#pragma unroll
  for (int i = 0; i < 8; i++) z[i] = 0.f;
  return z;
}

// ---------- data prep ----------
__global__ void gather_embed_kernel(const int* __restrict__ x,
                                    const float* __restrict__ emb,
                                    bf16* __restrict__ Xe) {
  int row = blockIdx.x;           // b*S+s
  int t = threadIdx.x;            // 0..319 (E=300 padded to 320)
  int id = x[row];
  float v = (t < 300) ? emb[(size_t)id * 300 + t] : 0.f;
  Xe[(size_t)row * 320 + t] = (bf16)v;
}

__global__ void cvt_pad_kernel(const float* __restrict__ src, bf16* __restrict__ dst,
                               int rows, int cols, int colsP) {
  int idx = blockIdx.x * blockDim.x + threadIdx.x;
  if (idx >= rows * colsP) return;
  int r = idx / colsP, c = idx % colsP;
  dst[idx] = (bf16)((c < cols) ? src[(size_t)r * cols + c] : 0.f);
}

__global__ void cvt_T_kernel(const float* __restrict__ src, bf16* __restrict__ dst,
                             int R, int C) {
  int idx = blockIdx.x * blockDim.x + threadIdx.x;
  if (idx >= R * C) return;
  int r = idx / C, c = idx % C;
  dst[(size_t)c * R + r] = (bf16)src[idx];
}

// ---------- generic bf16 WMMA GEMM:  C[M,N] = A[MxK] * Bw[NxK]^T + bias ----------
// 32x64 wave tile (2 M-subtiles x 4 N-subtiles).  K unrolled by 64 with two
// fragment sets loaded up front (copy-free ping-pong): set-B loads overlap
// set-A WMMAs.  M % 32 == 0, N % 64 == 0, K % 64 == 0.
__global__ __launch_bounds__(256, 1)
void gemm_bf16_kernel(const bf16* __restrict__ A, const bf16* __restrict__ Bw,
                      const float* __restrict__ bias, void* __restrict__ Cout,
                      int M, int N, int K, int outBf16) {
  int wave = blockIdx.x * (blockDim.x >> 5) + (threadIdx.x >> 5);
  int lane = threadIdx.x & 31;
  int l = lane & 15, hi = lane >> 4;
  int tilesN = N >> 6;
  int tilesM = M >> 5;
  if (wave >= tilesM * tilesN) return;
  int mBase = (wave / tilesN) << 5;
  int nBase = (wave % tilesN) << 6;

  v8f acc[2][4];
#pragma unroll
  for (int mi = 0; mi < 2; mi++)
#pragma unroll
    for (int t = 0; t < 4; t++) acc[mi][t] = vzero8();

  const bf16* aRow0 = A + (size_t)(mBase + l) * K;
  const bf16* aRow1 = A + (size_t)(mBase + 16 + l) * K;

#pragma unroll 1
  for (int k0 = 0; k0 < K; k0 += 64) {
    v16bf a0A = loadA_frag(aRow0 + k0, hi);
    v16bf a1A = loadA_frag(aRow1 + k0, hi);
    v16bf a0B = loadA_frag(aRow0 + k0 + 32, hi);
    v16bf a1B = loadA_frag(aRow1 + k0 + 32, hi);
    v16bf bA[4], bB[4];
#pragma unroll
    for (int t = 0; t < 4; t++) {
      const bf16* p = Bw + (size_t)(nBase + t * 16 + l) * K + k0;
      bA[t] = loadB_frag(p, hi);
      bB[t] = loadB_frag(p + 32, hi);
    }
#pragma unroll
    for (int t = 0; t < 4; t++) {
      acc[0][t] = wmma_bf16(a0A, bA[t], acc[0][t]);
      acc[1][t] = wmma_bf16(a1A, bA[t], acc[1][t]);
    }
#pragma unroll
    for (int t = 0; t < 4; t++) {
      acc[0][t] = wmma_bf16(a0B, bB[t], acc[0][t]);
      acc[1][t] = wmma_bf16(a1B, bB[t], acc[1][t]);
    }
  }

#pragma unroll
  for (int mi = 0; mi < 2; mi++) {
#pragma unroll
    for (int t = 0; t < 4; t++) {
      int col = nBase + t * 16 + l;
      float bv = bias ? bias[col] : 0.f;
#pragma unroll
      for (int r = 0; r < 8; r++) {
        int row = mBase + mi * 16 + r + 8 * hi;
        float v = acc[mi][t][r] + bv;
        if (outBf16) ((bf16*)Cout)[(size_t)row * N + col] = (bf16)v;
        else         ((float*)Cout)[(size_t)row * N + col] = v;
      }
    }
  }
}

// ---------- persistent LSTM direction kernel ----------
// grid = (4 M-tiles, 2 directions), block = 256 threads (8 waves).
// Per step: gates(16x1024) = h(16x256) @ Whh^T via WMMA, then fused gate math.
// Each wave owns gate columns [wave*128, wave*128+128), processed as two N=64
// passes with K unrolled by 64 (copy-free ping-pong, ~130 live VGPRs).
__global__ __launch_bounds__(256, 1)
void lstm_dir_kernel(const bf16* __restrict__ xgBase,
                     const bf16* __restrict__ whhBase,
                     bf16* __restrict__ hout) {
  const int S_ = 256, H_ = 256, G_ = 1024;
  int dir = blockIdx.y;
  int mBase = blockIdx.x * 16;
  const bf16* xg  = xgBase  + (size_t)dir * (16384ull * 1024ull);
  const bf16* Whh = whhBase + (size_t)dir * (1024 * 256);

  __shared__ bf16 hA[16][264];     // h state, A operand (row stride 528B = 33*16B)
  __shared__ bf16 gl[16][1024];    // gate staging

  int tid = threadIdx.x;
  int wave = tid >> 5, lane = tid & 31, l = lane & 15, hi = lane >> 4;

  for (int i = tid; i < 16 * 264; i += 256) ((bf16*)hA)[i] = (bf16)0.f;
  float c[16];
#pragma unroll
  for (int r = 0; r < 16; r++) c[r] = 0.f;
  __syncthreads();

  for (int s = 0; s < S_; s++) {
    int s_eff = dir ? (S_ - 1 - s) : s;

    // ---- WMMA phase ----
#pragma unroll 1
    for (int nh = 0; nh < 2; nh++) {
      int nBase2 = wave * 128 + nh * 64;
      v8f acc[4];
#pragma unroll
      for (int t = 0; t < 4; t++) acc[t] = vzero8();

#pragma unroll 1
      for (int k0 = 0; k0 < H_; k0 += 64) {
        v16bf aA = loadA_frag(&hA[l][k0], hi);
        v16bf aB = loadA_frag(&hA[l][k0 + 32], hi);
        v16bf bA[4], bB[4];
#pragma unroll
        for (int t = 0; t < 4; t++) {
          const bf16* p = Whh + (size_t)(nBase2 + t * 16 + l) * H_ + k0;
          bA[t] = loadB_frag(p, hi);
          bB[t] = loadB_frag(p + 32, hi);
        }
#pragma unroll
        for (int t = 0; t < 4; t++) acc[t] = wmma_bf16(aA, bA[t], acc[t]);
#pragma unroll
        for (int t = 0; t < 4; t++) acc[t] = wmma_bf16(aB, bB[t], acc[t]);
      }
#pragma unroll
      for (int t = 0; t < 4; t++) {
        int col = nBase2 + t * 16 + l;
#pragma unroll
        for (int r = 0; r < 8; r++) gl[r + 8 * hi][col] = (bf16)acc[t][r];
      }
    }
    __syncthreads();

    // ---- gate math: thread tid owns hidden unit j = tid for all 16 rows ----
    int j = tid;
    int s_nxt = dir ? (S_ - 2 - s) : (s + 1);
#pragma unroll 1
    for (int r = 0; r < 16; r++) {
      size_t xrow = ((size_t)(mBase + r) * S_ + s_eff) * G_;
      if (s + 1 < S_) {             // prefetch next timestep's gate pre-activations
        size_t xnxt = ((size_t)(mBase + r) * S_ + s_nxt) * G_;
        __builtin_prefetch(xg + xnxt + j, 0, 1);
        __builtin_prefetch(xg + xnxt + 512 + j, 0, 1);
      }
      float gi = (float)gl[r][j]        + (float)xg[xrow + j];
      float gf = (float)gl[r][256 + j]  + (float)xg[xrow + 256 + j];
      float gg = (float)gl[r][512 + j]  + (float)xg[xrow + 512 + j];
      float go = (float)gl[r][768 + j]  + (float)xg[xrow + 768 + j];
      float si = 1.f / (1.f + __expf(-gi));
      float sf = 1.f / (1.f + __expf(-gf));
      float so = 1.f / (1.f + __expf(-go));
      c[r] = sf * c[r] + si * tanhf(gg);
      float h = so * tanhf(c[r]);
      hA[r][j] = (bf16)h;
      hout[((size_t)(mBase + r) * S_ + s_eff) * 512 + dir * 256 + j] = (bf16)h;
    }
    __syncthreads();
  }
}

// ---------- positional MLP + residual LN (tiny, VALU) ----------
__global__ void pos_kernel(const float* __restrict__ pos,
                           const float* __restrict__ w1, const float* __restrict__ b1,
                           const float* __restrict__ w2, const float* __restrict__ b2,
                           const float* __restrict__ lg, const float* __restrict__ lb,
                           bf16* __restrict__ posout_bf) {
  __shared__ float pr[256];
  __shared__ float h1[256];
  __shared__ float red[256];
  int b = blockIdx.x, j = threadIdx.x;
  pr[j] = pos[b * 256 + j];
  __syncthreads();
  float acc = b1[j];
  for (int k = 0; k < 256; k++) acc += pr[k] * w1[k * 256 + j];
  h1[j] = fmaxf(acc, 0.f);
  __syncthreads();
  float acc2 = b2[j];
  for (int k = 0; k < 256; k++) acc2 += h1[k] * w2[k * 256 + j];
  float v = acc2 + pr[j];
  red[j] = v; __syncthreads();
  for (int st = 128; st > 0; st >>= 1) { if (j < st) red[j] += red[j + st]; __syncthreads(); }
  float mean = red[0] * (1.f / 256.f); __syncthreads();
  float d = v - mean; red[j] = d * d; __syncthreads();
  for (int st = 128; st > 0; st >>= 1) { if (j < st) red[j] += red[j + st]; __syncthreads(); }
  float var = red[0] * (1.f / 256.f);
  posout_bf[b * 256 + j] = (bf16)(d * rsqrtf(var + 1e-5f) * lg[j] + lb[j]);
}

// ---------- conditional LN (in place) + sentence reduction ----------
__global__ void cln_kernel(float* __restrict__ out3, const float* __restrict__ wMat,
                           const float* __restrict__ bMat, const float* __restrict__ alw,
                           const float* __restrict__ alb, float* __restrict__ sentence) {
  __shared__ float red[256];
  int row = blockIdx.x;            // b*S+s
  int b = row >> 8;
  int j = threadIdx.x;
  float v = out3[(size_t)row * 256 + j];
  red[j] = v; __syncthreads();
  for (int st = 128; st > 0; st >>= 1) { if (j < st) red[j] += red[j + st]; __syncthreads(); }
  float mean = red[0] * (1.f / 256.f); __syncthreads();
  float d = v - mean; red[j] = d * d; __syncthreads();
  for (int st = 128; st > 0; st >>= 1) { if (j < st) red[j] += red[j + st]; __syncthreads(); }
  float var = red[0] * (1.f / 256.f); __syncthreads();
  float o = d * rsqrtf(var + 1e-12f) * wMat[b * 256 + j] + bMat[b * 256 + j];
  out3[(size_t)row * 256 + j] = o;
  red[j] = o * alw[j]; __syncthreads();
  for (int st = 128; st > 0; st >>= 1) { if (j < st) red[j] += red[j + st]; __syncthreads(); }
  if (j == 0) sentence[row] = red[0] + alb[0];
}

// ---------- scalar attention + LN ----------
__global__ void att_kernel(const float* __restrict__ sent,
                           const float* wq, const float* bq, const float* wk,
                           const float* bk, const float* wv, const float* bv,
                           const float* lg, const float* lb, float* __restrict__ att) {
  __shared__ float sr[256];
  __shared__ float red[256];
  int b = blockIdx.x, j = threadIdx.x;
  sr[j] = sent[b * 256 + j];
  __syncthreads();
  float q = bq[j], k = bk[j], v = bv[j];
  for (int t = 0; t < 256; t++) {
    float s = sr[t];
    q += s * wq[t * 256 + j]; k += s * wk[t * 256 + j]; v += s * wv[t * 256 + j];
  }
  float sc = q * k * (1.f / 16.f);   // sqrt(256)=16
  red[j] = sc; __syncthreads();
  for (int st = 128; st > 0; st >>= 1) { if (j < st) red[j] = fmaxf(red[j], red[j + st]); __syncthreads(); }
  float mx = red[0]; __syncthreads();
  float e = __expf(sc - mx); red[j] = e; __syncthreads();
  for (int st = 128; st > 0; st >>= 1) { if (j < st) red[j] += red[j + st]; __syncthreads(); }
  float sum = red[0]; __syncthreads();
  float av = (e / sum) * v;
  red[j] = av; __syncthreads();
  for (int st = 128; st > 0; st >>= 1) { if (j < st) red[j] += red[j + st]; __syncthreads(); }
  float mean = red[0] * (1.f / 256.f); __syncthreads();
  float d = av - mean; red[j] = d * d; __syncthreads();
  for (int st = 128; st > 0; st >>= 1) { if (j < st) red[j] += red[j + st]; __syncthreads(); }
  float var = red[0] * (1.f / 256.f);
  att[b * 256 + j] = d * rsqrtf(var + 1e-5f) * lg[j] + lb[j];
}

// ---------- logits ----------
__global__ void logits_kernel(const float* __restrict__ out4, const float* __restrict__ att,
                              const float* __restrict__ ow, const float* __restrict__ ob,
                              float* __restrict__ logits) {
  int row = blockIdx.x;            // b*S+s  (att is (B,S) flattened the same way)
  int t = threadIdx.x;
  if (t < 13) {
    float scale = att[row];
    const float* r = out4 + (size_t)row * 256;
    float acc = 0.f;
    for (int k = 0; k < 256; k++) acc += r[k] * ow[k * 13 + t];
    logits[(size_t)row * 13 + t] = scale * acc + ob[t];
  }
}

// ---------- Viterbi ----------
__global__ void viterbi_kernel(const float* __restrict__ logits,
                               const float* __restrict__ cs, const float* __restrict__ ce,
                               const float* __restrict__ tr, int* __restrict__ hist,
                               float* __restrict__ tagsOut) {
  __shared__ float sc[16];
  __shared__ float ns[16];
  int b = blockIdx.x, t = threadIdx.x;   // 32 threads
  const float* em = logits + (size_t)b * 256 * 13;
  if (t < 13) sc[t] = cs[t] + em[t];
  __syncthreads();
  for (int s = 1; s < 256; s++) {
    if (t < 13) {
      float best = -1e30f; int bi = 0;
      for (int i = 0; i < 13; i++) {
        float v = sc[i] + tr[i * 13 + t];
        if (v > best) { best = v; bi = i; }
      }
      ns[t] = best + em[s * 13 + t];
      hist[((size_t)b * 255 + (s - 1)) * 13 + t] = bi;
    }
    __syncthreads();
    if (t < 13) sc[t] = ns[t];
    __syncthreads();
  }
  if (t == 0) {
    float best = -1e30f; int bi = 0;
    for (int i = 0; i < 13; i++) { float v = sc[i] + ce[i]; if (v > best) { best = v; bi = i; } }
    int tag = bi;
    tagsOut[b * 256 + 255] = (float)tag;
    for (int s = 254; s >= 0; s--) {
      tag = hist[((size_t)b * 255 + s) * 13 + tag];
      tagsOut[b * 256 + s] = (float)tag;
    }
  }
}

extern "C" void kernel_launch(void* const* d_in, const int* in_sizes, int n_in,
                              void* d_out, int out_size, void* d_ws, size_t ws_size,
                              hipStream_t stream) {
  (void)in_sizes; (void)n_in; (void)out_size; (void)ws_size;
  const int*   x     = (const int*)  d_in[0];
  const float* pos   = (const float*)d_in[1];
  const float* emb   = (const float*)d_in[2];
  const float* Wih0  = (const float*)d_in[3];
  const float* Whh0  = (const float*)d_in[4];
  const float* b0    = (const float*)d_in[5];
  const float* Wih1  = (const float*)d_in[6];
  const float* Whh1  = (const float*)d_in[7];
  const float* b1    = (const float*)d_in[8];
  const float* fc1w  = (const float*)d_in[9];
  const float* fc1b  = (const float*)d_in[10];
  const float* fc2w  = (const float*)d_in[11];
  const float* fc2b  = (const float*)d_in[12];
  const float* plg   = (const float*)d_in[13];
  const float* plb   = (const float*)d_in[14];
  const float* condw = (const float*)d_in[15];
  const float* condb = (const float*)d_in[16];
  const float* clnW  = (const float*)d_in[17];
  const float* clnB  = (const float*)d_in[18];
  const float* clnWd = (const float*)d_in[19];
  const float* clnBd = (const float*)d_in[20];
  const float* awq   = (const float*)d_in[21];
  const float* abq   = (const float*)d_in[22];
  const float* awk   = (const float*)d_in[23];
  const float* abk   = (const float*)d_in[24];
  const float* awv   = (const float*)d_in[25];
  const float* abv   = (const float*)d_in[26];
  const float* alg   = (const float*)d_in[27];
  const float* albn  = (const float*)d_in[28];
  const float* attlw = (const float*)d_in[29];
  const float* attlb = (const float*)d_in[30];
  const float* outw  = (const float*)d_in[31];
  const float* outb  = (const float*)d_in[32];
  const float* crfS  = (const float*)d_in[33];
  const float* crfE  = (const float*)d_in[34];
  const float* crfT  = (const float*)d_in[35];

  char* ws = (char*)d_ws;
  size_t off = 0;
  auto carve = [&](size_t bytes) -> char* {
    char* p = ws + off;
    off += (bytes + 255) & ~(size_t)255;
    return p;
  };
  bf16*  Xe      = (bf16*) carve(16384ull * 320 * 2);
  bf16*  wih0bf  = (bf16*) carve(2048ull * 320 * 2);
  bf16*  whh0bf  = (bf16*) carve(2048ull * 256 * 2);
  bf16*  wih1bf  = (bf16*) carve(2048ull * 512 * 2);
  bf16*  whh1bf  = (bf16*) carve(2048ull * 256 * 2);
  bf16*  condwT  = (bf16*) carve(256ull * 512 * 2);
  bf16*  clnwdbf = (bf16*) carve(65536ull * 2);
  bf16*  clnbdbf = (bf16*) carve(65536ull * 2);
  bf16*  posbf   = (bf16*) carve(16384ull * 2);
  bf16*  xg      = (bf16*) carve(2ull * 16384 * 1024 * 2);  // reused layer0 -> layer1
  bf16*  h0cat   = (bf16*) carve(16384ull * 512 * 2);
  bf16*  h1cat   = (bf16*) carve(16384ull * 512 * 2);
  float* out3    = (float*)carve(16384ull * 256 * 4);
  float* wMat    = (float*)carve(64ull * 256 * 4);
  float* bMat    = (float*)carve(64ull * 256 * 4);
  float* sentence= (float*)carve(64ull * 256 * 4);
  float* attB    = (float*)carve(64ull * 256 * 4);
  int*   hist    = (int*)  carve(64ull * 255 * 13 * 4);

  // data prep
  gather_embed_kernel<<<16384, 320, 0, stream>>>(x, emb, Xe);
  cvt_pad_kernel<<<(2048 * 320 + 255) / 256, 256, 0, stream>>>(Wih0, wih0bf, 2048, 300, 320);
  cvt_pad_kernel<<<(2048 * 256 + 255) / 256, 256, 0, stream>>>(Whh0, whh0bf, 2048, 256, 256);
  cvt_pad_kernel<<<(2048 * 512 + 255) / 256, 256, 0, stream>>>(Wih1, wih1bf, 2048, 512, 512);
  cvt_pad_kernel<<<(2048 * 256 + 255) / 256, 256, 0, stream>>>(Whh1, whh1bf, 2048, 256, 256);
  cvt_T_kernel<<<(512 * 256 + 255) / 256, 256, 0, stream>>>(condw, condwT, 512, 256);
  cvt_pad_kernel<<<(65536 + 255) / 256, 256, 0, stream>>>(clnWd, clnwdbf, 256, 256, 256);
  cvt_pad_kernel<<<(65536 + 255) / 256, 256, 0, stream>>>(clnBd, clnbdbf, 256, 256, 256);

  // LSTM layer 0: input projections (WMMA) + recurrence (persistent WMMA)
  for (int dir = 0; dir < 2; dir++) {
    int tiles = (16384 >> 5) * (1024 >> 6);
    gemm_bf16_kernel<<<(tiles + 7) / 8, 256, 0, stream>>>(
        Xe, wih0bf + (size_t)dir * 1024 * 320, b0 + dir * 1024,
        xg + (size_t)dir * 16384 * 1024, 16384, 1024, 320, 1);
  }
  lstm_dir_kernel<<<dim3(4, 2), 256, 0, stream>>>(xg, whh0bf, h0cat);

  // LSTM layer 1
  for (int dir = 0; dir < 2; dir++) {
    int tiles = (16384 >> 5) * (1024 >> 6);
    gemm_bf16_kernel<<<(tiles + 7) / 8, 256, 0, stream>>>(
        h0cat, wih1bf + (size_t)dir * 1024 * 512, b1 + dir * 1024,
        xg + (size_t)dir * 16384 * 1024, 16384, 1024, 512, 1);
  }
  lstm_dir_kernel<<<dim3(4, 2), 256, 0, stream>>>(xg, whh1bf, h1cat);

  // cond projection (WMMA)
  {
    int tiles = (16384 >> 5) * (256 >> 6);
    gemm_bf16_kernel<<<(tiles + 7) / 8, 256, 0, stream>>>(h1cat, condwT, condb, out3,
                                                          16384, 256, 512, 0);
  }

  // positional net, conditional LN params (WMMA), cLN+sentence, attention
  pos_kernel<<<64, 256, 0, stream>>>(pos, fc1w, fc1b, fc2w, fc2b, plg, plb, posbf);
  {
    int tiles = (64 >> 5) * (256 >> 6);
    gemm_bf16_kernel<<<(tiles + 7) / 8, 256, 0, stream>>>(posbf, clnwdbf, clnW, wMat,
                                                          64, 256, 256, 0);
    gemm_bf16_kernel<<<(tiles + 7) / 8, 256, 0, stream>>>(posbf, clnbdbf, clnB, bMat,
                                                          64, 256, 256, 0);
  }
  cln_kernel<<<16384, 256, 0, stream>>>(out3, wMat, bMat, attlw, attlb, sentence);
  att_kernel<<<64, 256, 0, stream>>>(sentence, awq, abq, awk, abk, awv, abv, alg, albn, attB);

  // logits + Viterbi into d_out (logits fp32, then tags as float)
  float* logits = (float*)d_out;
  float* tags = logits + 16384 * 13;
  logits_kernel<<<16384, 32, 0, stream>>>(out3, attB, outw, outb, logits);
  viterbi_kernel<<<64, 32, 0, stream>>>(logits, crfS, crfE, crfT, hist, tags);
}
